// UnifiedMultiHeadAttention_50534585205451
// MI455X (gfx1250) — matrix operand order repaired and checked by reading, not compiled, not run on hardware
//
#include <hip/hip_runtime.h>
#include <hip/hip_bf16.h>

// ---------------------------------------------------------------------------
// Problem constants (B=4, T=2048, C=768, H=12, hd=64)
// ---------------------------------------------------------------------------
#define BB   4
#define TT   2048
#define CC   768
#define HH   12
#define HD   64
#define BT   (BB * TT)          // 8192
#define C3   (3 * CC)           // 2304

typedef __attribute__((ext_vector_type(16))) __bf16   v16bf;
typedef __attribute__((ext_vector_type(8)))  float    v8f;
typedef __attribute__((ext_vector_type(8)))  unsigned int v8u;

// fp32 -> bf16 round-to-nearest-even
__device__ __forceinline__ unsigned short f2bf(float f) {
    unsigned int u = __builtin_bit_cast(unsigned int, f);
    u += 0x7fffu + ((u >> 16) & 1u);
    return (unsigned short)(u >> 16);
}

__device__ __forceinline__ v8f wmma_bf16(v16bf a, v16bf b, v8f c) {
    return __builtin_amdgcn_wmma_f32_16x16x32_bf16(
        false, a, false, b, (short)0, c, false, false);
}

// A fragment (16Mx32K bf16) from row-major [M][K] storage, ld in elements.
// ISA layout: lanes 0-15 -> M=lane, K pairs {0..7,16..23}; lanes 16-31 ->
// same M, K pairs {8..15,24..31}.
__device__ __forceinline__ v16bf load_frag_a(const unsigned short* base, int ld, int lane) {
    const int half = lane >> 4;
    const unsigned short* rowp = base + (lane & 15) * ld;
    v8u u;
#pragma unroll
    for (int v = 0; v < 8; ++v) {
        const int k = ((v & 4) << 2) + (half << 3) + ((v & 3) << 1);
        u[v] = *(const unsigned int*)(rowp + k);
    }
    return __builtin_bit_cast(v16bf, u);
}

// B fragment (32Kx16N bf16) from N-major ("transposed") [N][K] storage.
// ISA layout: lanes 0-15 -> N=lane, K pairs {0..7,16..23}; lanes 16-31 ->
// N=lane-16, K pairs {8..15,24..31}.
__device__ __forceinline__ v16bf load_frag_b(const unsigned short* baseT, int ld, int lane) {
    const int half = lane >> 4;
    const unsigned short* rowp = baseT + (lane & 15) * ld;
    v8u u;
#pragma unroll
    for (int v = 0; v < 8; ++v)
        u[v] = *(const unsigned int*)(rowp + ((v & 4) << 2) + (half << 3) + ((v & 3) << 1));
    return __builtin_bit_cast(v16bf, u);
}

// 16-lane (half-wave) reductions: rows of the 16x16 C/D tile live in one half.
__device__ __forceinline__ float rmax16(float x) {
#pragma unroll
    for (int m = 1; m < 16; m <<= 1) x = fmaxf(x, __shfl_xor(x, m, 32));
    return x;
}
__device__ __forceinline__ float rsum16(float x) {
#pragma unroll
    for (int m = 1; m < 16; m <<= 1) x += __shfl_xor(x, m, 32);
    return x;
}

// ---------------------------------------------------------------------------
// Conversion kernels (one-time fp32 -> bf16, weights transposed to [out][in])
// ---------------------------------------------------------------------------
__global__ void cvt_kernel(const float* __restrict__ src,
                           unsigned short* __restrict__ dst, int n) {
    int i = blockIdx.x * blockDim.x + threadIdx.x;
    if (i < n) dst[i] = f2bf(src[i]);
}

__global__ void transpose_cvt_kernel(const float* __restrict__ w,
                                     unsigned short* __restrict__ wT,
                                     int rows, int cols) {   // w[rows][cols] -> wT[cols][rows]
    int i = blockIdx.x * blockDim.x + threadIdx.x;
    if (i < rows * cols) {
        int r = i / cols, c = i - r * cols;
        wT[c * rows + r] = f2bf(w[i]);
    }
}

// ---------------------------------------------------------------------------
// QKV projection: qkv = x @ w_attn + b_attn, scattered into
//   q,k : bf16 [B,H,T,hd]     vT : bf16 [B,H,hd,T]  (pre-transposed for P@V)
// Wave tile: 32(M) x 64(N), K-loop over C in steps of 32.
// ---------------------------------------------------------------------------
__global__ void qkv_gemm_kernel(const unsigned short* __restrict__ xb,
                                const unsigned short* __restrict__ wT,
                                const float* __restrict__ bias,
                                unsigned short* __restrict__ q,
                                unsigned short* __restrict__ k,
                                unsigned short* __restrict__ vT) {
    const int wave = blockIdx.x * (blockDim.x >> 5) + (threadIdx.x >> 5);
    const int lane = threadIdx.x & 31;
    const int nT = C3 / 64;                   // 36
    const int mt = wave / nT, nt = wave - mt * nT;
    const int m0 = mt * 32, n0 = nt * 64;

    v8f acc[8] = {};                          // [i2*4 + j]
    for (int k0 = 0; k0 < CC; k0 += 32) {
        v16bf a0 = load_frag_a(xb + (size_t)m0 * CC + k0, CC, lane);
        v16bf a1 = load_frag_a(xb + (size_t)(m0 + 16) * CC + k0, CC, lane);
#pragma unroll
        for (int j = 0; j < 4; ++j) {
            v16bf b = load_frag_b(wT + (size_t)(n0 + 16 * j) * CC + k0, CC, lane);
            acc[j]     = wmma_bf16(a0, b, acc[j]);
            acc[4 + j] = wmma_bf16(a1, b, acc[4 + j]);
        }
    }

    const int nloc = lane & 15, rbase = (lane >> 4) * 8;
#pragma unroll
    for (int i2 = 0; i2 < 2; ++i2)
#pragma unroll
        for (int j = 0; j < 4; ++j) {
            v8f a = acc[i2 * 4 + j];
#pragma unroll
            for (int i = 0; i < 8; ++i) {
                const int m = m0 + i2 * 16 + rbase + i;
                const int n = n0 + 16 * j + nloc;
                const unsigned short h = f2bf(a[i] + bias[n]);
                const int which = n / CC, nn = n - which * CC;
                const int hh = nn >> 6, d = nn & 63;
                const int b_ = m >> 11, t = m & (TT - 1);
                const int bh = b_ * HH + hh;
                if (which == 0)      q [((size_t)bh * TT + t) * HD + d] = h;
                else if (which == 1) k [((size_t)bh * TT + t) * HD + d] = h;
                else                 vT[((size_t)bh * HD + d) * TT + t] = h;
            }
        }
}

// ---------------------------------------------------------------------------
// Causal flash attention. One wave per (b,h, 16-row Q tile).
// S computed 16x32 per step (2x2 WMMA), online softmax in fp32, P staged
// through a wave-private LDS region to re-shape C-layout -> A-fragment,
// then 4 WMMAs of P @ V^T into fp32 output accumulators.
// ---------------------------------------------------------------------------
__global__ void attn_kernel(const unsigned short* __restrict__ q,
                            const unsigned short* __restrict__ k,
                            const unsigned short* __restrict__ vT,
                            unsigned short* __restrict__ yb) {
    __shared__ unsigned short smem[8 * 16 * 32];       // 8 waves x (16x32 bf16 P tile)
    const int wave = blockIdx.x * (blockDim.x >> 5) + (threadIdx.x >> 5);
    const int lane = threadIdx.x & 31;
    const int woff = (threadIdx.x >> 5) * (16 * 32);

    const int tTiles = TT / 16;                        // 128
    const int bh = wave / tTiles;
    const int t0 = (wave - bh * tTiles) * 16;

    const unsigned short* qbase = q  + ((size_t)bh * TT + t0) * HD;
    const unsigned short* kbase = k  + (size_t)bh * TT * HD;
    const unsigned short* vbase = vT + (size_t)bh * HD * TT;

    const v16bf qa0 = load_frag_a(qbase, HD, lane);        // K = 0..31
    const v16bf qa1 = load_frag_a(qbase + 32, HD, lane);   // K = 32..63

    const int nloc = lane & 15, rbase = (lane >> 4) * 8;
    const float scale = 0.125f;                        // 1/sqrt(64)

    v8f   o[4] = {};
    float mrow[8], lrow[8];
#pragma unroll
    for (int i = 0; i < 8; ++i) { mrow[i] = -1e30f; lrow[i] = 0.0f; }

    for (int s0 = 0; s0 < t0 + 16; s0 += 32) {
        // hint the next K/V chunk toward the caches
        __builtin_prefetch(kbase + (size_t)(s0 + 32) * HD, 0, 1);

        // ---- S = Q @ K^T for columns [s0, s0+32) ----
        v8f sa0 = {}, sa1 = {};
        {
            v16bf b0 = load_frag_b(kbase + (size_t)s0 * HD, HD, lane);
            v16bf b1 = load_frag_b(kbase + (size_t)s0 * HD + 32, HD, lane);
            sa0 = wmma_bf16(qa0, b0, sa0);
            sa0 = wmma_bf16(qa1, b1, sa0);
        }
        {
            v16bf b0 = load_frag_b(kbase + (size_t)(s0 + 16) * HD, HD, lane);
            v16bf b1 = load_frag_b(kbase + (size_t)(s0 + 16) * HD + 32, HD, lane);
            sa1 = wmma_bf16(qa0, b0, sa1);
            sa1 = wmma_bf16(qa1, b1, sa1);
        }

        // ---- online softmax over the 32 new columns ----
        float p0[8], p1[8], alpha[8];
#pragma unroll
        for (int i = 0; i < 8; ++i) {
            const int r  = t0 + rbase + i;             // global row
            const int c0 = s0 + nloc;
            const int c1 = s0 + 16 + nloc;
            float s0v = (c0 > r) ? -1e30f : sa0[i] * scale;
            float s1v = (c1 > r) ? -1e30f : sa1[i] * scale;
            float mx   = rmax16(fmaxf(s0v, s1v));
            float mnew = fmaxf(mrow[i], mx);
            float a    = __expf(mrow[i] - mnew);
            float e0   = __expf(s0v - mnew);
            float e1   = __expf(s1v - mnew);
            lrow[i] = lrow[i] * a + rsum16(e0 + e1);
            mrow[i] = mnew;
            alpha[i] = a; p0[i] = e0; p1[i] = e1;
        }
#pragma unroll
        for (int j = 0; j < 4; ++j)
#pragma unroll
            for (int i = 0; i < 8; ++i) o[j][i] *= alpha[i];

        // ---- stage P (16x32) through LDS: C-layout regs -> A fragment ----
#pragma unroll
        for (int i = 0; i < 8; ++i) {
            smem[woff + (rbase + i) * 32 + nloc]      = f2bf(p0[i]);
            smem[woff + (rbase + i) * 32 + 16 + nloc] = f2bf(p1[i]);
        }
        v8u u;
        {
            const int half = lane >> 4;
            const int mrow_ = lane & 15;
#pragma unroll
            for (int v = 0; v < 8; ++v) {
                const int kk = ((v & 4) << 2) + (half << 3) + ((v & 3) << 1);
                u[v] = *(const unsigned int*)&smem[woff + mrow_ * 32 + kk];
            }
        }
        const v16bf pa = __builtin_bit_cast(v16bf, u);

        // ---- O += P @ V   (V^T is N-major storage for B fragments) ----
#pragma unroll
        for (int j = 0; j < 4; ++j) {
            v16bf vb = load_frag_b(vbase + (size_t)(16 * j) * TT + s0, TT, lane);
            o[j] = wmma_bf16(pa, vb, o[j]);
        }
    }

    // ---- normalize and write y as bf16 [B,T,C] ----
    float inv[8];
#pragma unroll
    for (int i = 0; i < 8; ++i) inv[i] = 1.0f / lrow[i];

    const int b_ = bh / HH, h_ = bh - b_ * HH;
#pragma unroll
    for (int j = 0; j < 4; ++j)
#pragma unroll
        for (int i = 0; i < 8; ++i) {
            const int t = t0 + rbase + i;
            const int c = h_ * HD + 16 * j + nloc;
            yb[((size_t)b_ * TT + t) * CC + c] = f2bf(o[j][i] * inv[i]);
        }
}

// ---------------------------------------------------------------------------
// Output projection: out = y @ w_proj + b_proj (fp32 result)
// ---------------------------------------------------------------------------
__global__ void proj_gemm_kernel(const unsigned short* __restrict__ yb,
                                 const unsigned short* __restrict__ wT,
                                 const float* __restrict__ bias,
                                 float* __restrict__ out) {
    const int wave = blockIdx.x * (blockDim.x >> 5) + (threadIdx.x >> 5);
    const int lane = threadIdx.x & 31;
    const int nT = CC / 64;                   // 12
    const int mt = wave / nT, nt = wave - mt * nT;
    const int m0 = mt * 32, n0 = nt * 64;

    v8f acc[8] = {};
    for (int k0 = 0; k0 < CC; k0 += 32) {
        v16bf a0 = load_frag_a(yb + (size_t)m0 * CC + k0, CC, lane);
        v16bf a1 = load_frag_a(yb + (size_t)(m0 + 16) * CC + k0, CC, lane);
#pragma unroll
        for (int j = 0; j < 4; ++j) {
            v16bf b = load_frag_b(wT + (size_t)(n0 + 16 * j) * CC + k0, CC, lane);
            acc[j]     = wmma_bf16(a0, b, acc[j]);
            acc[4 + j] = wmma_bf16(a1, b, acc[4 + j]);
        }
    }

    const int nloc = lane & 15, rbase = (lane >> 4) * 8;
#pragma unroll
    for (int i2 = 0; i2 < 2; ++i2)
#pragma unroll
        for (int j = 0; j < 4; ++j) {
            v8f a = acc[i2 * 4 + j];
#pragma unroll
            for (int i = 0; i < 8; ++i) {
                const int m = m0 + i2 * 16 + rbase + i;
                const int n = n0 + 16 * j + nloc;
                out[(size_t)m * CC + n] = a[i] + bias[n];
            }
        }
}

// ---------------------------------------------------------------------------
// Host launch
// ---------------------------------------------------------------------------
extern "C" void kernel_launch(void* const* d_in, const int* in_sizes, int n_in,
                              void* d_out, int out_size, void* d_ws, size_t ws_size,
                              hipStream_t stream) {
    const float* x      = (const float*)d_in[0];
    const float* w_attn = (const float*)d_in[1];
    const float* b_attn = (const float*)d_in[2];
    const float* w_proj = (const float*)d_in[3];
    const float* b_proj = (const float*)d_in[4];
    float* out = (float*)d_out;

    char* ws = (char*)d_ws;
    const size_t SZ_X  = (size_t)BT * CC * 2;      // 12.6 MB (bf16)
    const size_t SZ_WA = (size_t)C3 * CC * 2;      //  3.5 MB
    const size_t SZ_WP = (size_t)CC * CC * 2;      //  1.2 MB
    unsigned short* xb     = (unsigned short*)(ws);                    // also reused as yb
    unsigned short* wattnT = (unsigned short*)(ws + SZ_X);
    unsigned short* wprojT = (unsigned short*)(ws + SZ_X + SZ_WA);
    unsigned short* qb     = (unsigned short*)(ws + SZ_X + SZ_WA + SZ_WP);
    unsigned short* kb     = (unsigned short*)(ws + SZ_X + SZ_WA + SZ_WP + SZ_X);
    unsigned short* vTb    = (unsigned short*)(ws + SZ_X + SZ_WA + SZ_WP + 2 * SZ_X);
    unsigned short* yb     = xb;   // x-bf16 dead after QKV GEMM; reuse for y

    // 1) fp32 -> bf16 conversions / weight transposes
    {
        int n = BT * CC;
        cvt_kernel<<<(n + 255) / 256, 256, 0, stream>>>(x, xb, n);
    }
    {
        int n = CC * C3;
        transpose_cvt_kernel<<<(n + 255) / 256, 256, 0, stream>>>(w_attn, wattnT, CC, C3);
    }
    {
        int n = CC * CC;
        transpose_cvt_kernel<<<(n + 255) / 256, 256, 0, stream>>>(w_proj, wprojT, CC, CC);
    }

    // 2) QKV projection: (BT/32)*(C3/64) = 9216 waves -> 1152 blocks of 8 waves
    qkv_gemm_kernel<<<(BT / 32) * (C3 / 64) / 8, 256, 0, stream>>>(
        xb, wattnT, b_attn, qb, kb, vTb);

    // 3) Attention: B*H*(T/16) = 6144 waves -> 768 blocks
    attn_kernel<<<BB * HH * (TT / 16) / 8, 256, 0, stream>>>(qb, kb, vTb, yb);

    // 4) Output projection: (BT/32)*(CC/64) = 3072 waves -> 384 blocks
    proj_gemm_kernel<<<(BT / 32) * (CC / 64) / 8, 256, 0, stream>>>(
        yb, wprojT, b_proj, out);
}
